// LightningDecoderLayer_7181185319150
// MI455X (gfx1250) — matrix operand order repaired and checked by reading, not compiled, not run on hardware
//
#include <hip/hip_runtime.h>
#include <hip/hip_bf16.h>
#include <math.h>

// Problem constants (match reference)
#define Bb 2
#define Tt 2048
#define Cc 1024
#define Hh 16
#define Dd 64
#define KK 4
#define BT (Bb*Tt)          // 4096 rows

typedef __attribute__((ext_vector_type(16))) __bf16 v16bf;
typedef __attribute__((ext_vector_type(8)))  __bf16 v8bf;
typedef __attribute__((ext_vector_type(8)))  float  v8f;

union BF8 { uint4 u4; v8bf v; };

__device__ __forceinline__ unsigned short f2bf(float f){
  unsigned u = __float_as_uint(f);
  u += 0x7FFFu + ((u >> 16) & 1u);           // round-to-nearest-even
  return (unsigned short)(u >> 16);
}
__device__ __forceinline__ float sigm(float x){ return 1.0f/(1.0f+expf(-x)); }

// ---------------- cast f32 -> bf16 ----------------
__global__ __launch_bounds__(256) void cast_bf16_kernel(const float* __restrict__ in,
                                                        unsigned short* __restrict__ out, int n){
  for (int i = blockIdx.x*blockDim.x + threadIdx.x; i < n; i += gridDim.x*blockDim.x)
    out[i] = f2bf(in[i]);
}

// ---------------- WMMA GEMM: C[M,N] = A[M,K] * Bw[N,K]^T (all bf16 in, f32 out) -------------
// Block: 256 threads = 8 waves arranged 4(M) x 2(N); wave tile 32x32 (2x2 wmma frags).
// Block tile 128(M) x 64(N). Requires M%128==0, N%64==0, K%32==0.
__global__ __launch_bounds__(256) void gemm_bf16_kernel(const unsigned short* __restrict__ A,
                                                        const unsigned short* __restrict__ Bw,
                                                        float* __restrict__ Cout,
                                                        int M, int N, int K){
  const int lane = threadIdx.x & 31;
  const int wave = threadIdx.x >> 5;
  const int wm   = wave >> 1;          // 0..3
  const int wn   = wave & 1;           // 0..1
  const int m_base = blockIdx.y*128 + wm*32;
  const int n_base = blockIdx.x*64  + wn*32;
  const int lm  = lane & 15;
  const int lhi = lane >> 4;           // 0 or 1
  const int kbA = lhi*8;               // A: lanes>=16 carry K+8 slice
  const int kbB = lhi*8;               // B: lanes>=16 carry K+8 slice

  v8f acc[2][2] = {};

  for (int kk = 0; kk < K; kk += 32) {
    v16bf afr[2], bfr[2];
    #pragma unroll
    for (int i = 0; i < 2; i++) {
      const size_t row = (size_t)(m_base + i*16 + lm);
      BF8 lo, hi;
      lo.u4 = *(const uint4*)(A + row*K + kk + kbA);        // K = kb..kb+7
      hi.u4 = *(const uint4*)(A + row*K + kk + 16 + kbA);   // K = 16+kb..16+kb+7
      #pragma unroll
      for (int e = 0; e < 8; e++) { afr[i][e] = lo.v[e]; afr[i][8+e] = hi.v[e]; }
    }
    #pragma unroll
    for (int j = 0; j < 2; j++) {
      const size_t col = (size_t)(n_base + j*16 + lm);
      BF8 lo, hi;
      lo.u4 = *(const uint4*)(Bw + col*K + kk + kbB);       // K = kb..kb+7
      hi.u4 = *(const uint4*)(Bw + col*K + kk + 16 + kbB);  // K = 16+kb..16+kb+7
      #pragma unroll
      for (int e = 0; e < 8; e++) { bfr[j][e] = lo.v[e]; bfr[j][8+e] = hi.v[e]; }
    }
    #pragma unroll
    for (int i = 0; i < 2; i++)
      #pragma unroll
      for (int j = 0; j < 2; j++)
        acc[i][j] = __builtin_amdgcn_wmma_f32_16x16x32_bf16(
            false, afr[i], false, bfr[j], (short)0, acc[i][j], false, false);
  }

  #pragma unroll
  for (int i = 0; i < 2; i++)
    #pragma unroll
    for (int j = 0; j < 2; j++) {
      const int col = n_base + j*16 + lm;
      #pragma unroll
      for (int r = 0; r < 8; r++) {
        const int row = m_base + i*16 + lhi*8 + r;
        Cout[(size_t)row*N + col] = acc[i][j][r];
      }
    }
}

// ---------------- depthwise causal conv (K=4) + SiLU ----------------
__global__ __launch_bounds__(256) void conv_silu_kernel(const float* __restrict__ xin,
                                                        const float* __restrict__ w,
                                                        const float* __restrict__ bias,
                                                        float* __restrict__ y){
  const int total = BT*Cc;
  for (int i = blockIdx.x*blockDim.x + threadIdx.x; i < total; i += gridDim.x*blockDim.x) {
    const int c  = i % Cc;
    const int t  = (i / Cc) % Tt;
    const int b  = i / (Cc*Tt);
    float acc = bias[c];
    #pragma unroll
    for (int j = 0; j < KK; j++) {
      const int tt = t - (KK-1) + j;
      if (tt >= 0) acc += w[c*KK + j] * xin[((size_t)(b*Tt + tt))*Cc + c];
    }
    y[i] = acc * sigm(acc);   // silu
  }
}

// ---------------- RoPE (YaRN-style per reference) + L2 norm, in place ----------------
// One wave per (b,t,h) group of 64 values; lane d handles pair (2d, 2d+1) -> (d, d+32).
__global__ __launch_bounds__(256) void rope_l2norm_kernel(float* __restrict__ buf){
  const int wave = threadIdx.x >> 5;
  const int lane = threadIdx.x & 31;
  const int gid  = blockIdx.x*8 + wave;         // (b*T + t)*H + h
  if (gid >= BT*Hh) return;
  const int t = (gid / Hh) % Tt;
  const size_t base = (size_t)gid * Dd;

  const float x1 = buf[base + 2*lane];
  const float x2 = buf[base + 2*lane + 1];

  const int   fi = (2*lane) & 31;               // freq index used at emb slot 2d
  const float ar = (float)fi / 32.0f;
  const float scaled_base = 10000.0f * powf(32.0f, 64.0f/62.0f);
  const float inv_freq   = powf(scaled_base, -ar);
  const float freq_extra = powf(10000.0f, -ar);
  const float wavelen = 2.0f * 3.14159265358979323846f / freq_extra;
  float msc = (wavelen - 1.0f) / 31.0f;
  msc = fminf(fmaxf(msc, 0.0f), 1.0f);
  const float scale = 1.0f + 31.0f * msc;
  const float ang = ((float)t / scale) * inv_freq;
  const float ce = cosf(ang), se = sinf(ang);

  const float o1 = x1*ce - x2*se;
  const float o2 = x1*se + x2*ce;

  float ss = o1*o1 + o2*o2;
  #pragma unroll
  for (int off = 16; off > 0; off >>= 1) ss += __shfl_xor(ss, off, 32);
  const float nrm = sqrtf(ss);
  const float r = 1.0f / fmaxf(nrm, 1e-12f);

  buf[base + lane]      = o1 * r;
  buf[base + 32 + lane] = o2 * r;
}

// ---------------- alpha/beta projections (N=16 tiny GEMVs, full f32) ----------------
__global__ __launch_bounds__(64) void proj_ab_kernel(const float* __restrict__ x,
                                                     const float* __restrict__ Wb,
                                                     const float* __restrict__ bb,
                                                     const float* __restrict__ Wgk,
                                                     const float* __restrict__ bgk,
                                                     const float* __restrict__ A_log,
                                                     const float* __restrict__ dt_bias,
                                                     float* __restrict__ alpha,
                                                     float* __restrict__ beta){
  __shared__ float xs[Cc];
  const int bt = blockIdx.x;                    // 0..BT-1
  for (int i = threadIdx.x; i < Cc; i += 64) xs[i] = x[(size_t)bt*Cc + i];
  __syncthreads();
  const int tid = threadIdx.x;
  if (tid < Hh) {
    const int h = tid;
    float acc = 0.0f;
    for (int c = 0; c < Cc; c++) acc += xs[c] * Wb[h*Cc + c];
    beta[bt*Hh + h] = sigm(acc + bb[h]);
  } else if (tid < 2*Hh) {
    const int h = tid - Hh;
    float acc = 0.0f;
    for (int c = 0; c < Cc; c++) acc += xs[c] * Wgk[h*Cc + c];
    const float z = acc + bgk[h] + dt_bias[h];
    const float sp = (z > 20.0f) ? z : log1pf(expf(z));   // softplus
    alpha[bt*Hh + h] = expf(-expf(A_log[h]) * sp);
  }
}

// ---------------- gated delta-rule recurrent scan ----------------
// One block of 64 threads per (b,h); thread v holds S[v][0..63] in VGPRs.
__global__ __launch_bounds__(64) void scan_kernel(const float* __restrict__ q,
                                                  const float* __restrict__ k,
                                                  const float* __restrict__ v,
                                                  const float* __restrict__ alpha,
                                                  const float* __restrict__ beta,
                                                  const float* __restrict__ Dp,
                                                  float* __restrict__ o){
  const int bh = blockIdx.x;                    // 0..B*H-1
  const int b  = bh / Hh;
  const int h  = bh % Hh;
  const int tid = threadIdx.x;                  // 0..63 = v index
  const float dph = Dp[h];

  float S[Dd];
  #pragma unroll
  for (int i = 0; i < Dd; i++) S[i] = 0.0f;

  __shared__ float sh[3*Dd];                    // k | q | v

  for (int t = 0; t < Tt; t++) {
    const size_t base = ((size_t)(b*Tt + t)*Hh + h) * Dd;
    sh[tid]        = k[base + tid];
    sh[Dd + tid]   = q[base + tid];
    sh[2*Dd + tid] = v[base + tid];
    __syncthreads();

    const float a   = alpha[(b*Tt + t)*Hh + h];
    const float bt_ = beta [(b*Tt + t)*Hh + h];

    float Sk = 0.0f;
    #pragma unroll
    for (int kk = 0; kk < Dd; kk++) Sk += S[kk] * sh[kk];

    const float vv = sh[2*Dd + tid];
    float oo = 0.0f;
    #pragma unroll
    for (int kk = 0; kk < Dd; kk++) {
      const float kv = sh[kk];
      const float s  = a * (S[kk] - bt_ * Sk * kv) + bt_ * vv * kv;
      S[kk] = s;
      oo += s * sh[Dd + kk];
    }
    o[base + tid] = oo + dph * vv;              // fold +Dp*v here
    __syncthreads();
  }
}

// ---------------- RMSNorm + SiLU + gate, write bf16 for final GEMM ----------------
__global__ __launch_bounds__(256) void post_kernel(const float* __restrict__ o,
                                                   const float* __restrict__ g,
                                                   const float* __restrict__ w,
                                                   unsigned short* __restrict__ obf){
  const int wave = threadIdx.x >> 5;
  const int lane = threadIdx.x & 31;
  const int gid  = blockIdx.x*8 + wave;         // (b,t,h)
  if (gid >= BT*Hh) return;
  const size_t base = (size_t)gid * Dd;

  const float e0 = o[base + lane];
  const float e1 = o[base + 32 + lane];
  float ss = e0*e0 + e1*e1;
  #pragma unroll
  for (int off = 16; off > 0; off >>= 1) ss += __shfl_xor(ss, off, 32);
  const float r = rsqrtf(ss / 64.0f + 1e-6f);

  const float n0 = w[lane]      * (e0 * r);
  const float n1 = w[lane + 32] * (e1 * r);
  const float s0 = n0 * sigm(n0);
  const float s1 = n1 * sigm(n1);
  obf[base + lane]      = f2bf(g[base + lane]      * s0);
  obf[base + 32 + lane] = f2bf(g[base + 32 + lane] * s1);
}

// =====================================================================================
extern "C" void kernel_launch(void* const* d_in, const int* in_sizes, int n_in,
                              void* d_out, int out_size, void* d_ws, size_t ws_size,
                              hipStream_t stream) {
  (void)in_sizes; (void)n_in; (void)out_size; (void)ws_size;
  const float* x       = (const float*)d_in[0];
  const float* Wq      = (const float*)d_in[1];
  const float* Wk      = (const float*)d_in[2];
  const float* Wv      = (const float*)d_in[3];
  const float* Wg      = (const float*)d_in[4];
  const float* Wo      = (const float*)d_in[5];
  const float* Wb      = (const float*)d_in[6];
  const float* bb      = (const float*)d_in[7];
  const float* Wgk     = (const float*)d_in[8];
  const float* bgk     = (const float*)d_in[9];
  const float* cqw     = (const float*)d_in[10];
  const float* cqb     = (const float*)d_in[11];
  const float* ckw     = (const float*)d_in[12];
  const float* ckb     = (const float*)d_in[13];
  const float* cvw     = (const float*)d_in[14];
  const float* cvb     = (const float*)d_in[15];
  const float* A_log   = (const float*)d_in[16];
  const float* Dp      = (const float*)d_in[17];
  const float* dt_bias = (const float*)d_in[18];
  const float* onw     = (const float*)d_in[19];
  float* out = (float*)d_out;

  // -------- workspace layout --------
  unsigned char* ws = (unsigned char*)d_ws;
  size_t off = 0;
  auto alloc = [&](size_t bytes) -> void* {
    void* p = ws + off;
    off += (bytes + 255) & ~(size_t)255;
    return p;
  };
  unsigned short* xbf  = (unsigned short*)alloc((size_t)BT*Cc*2);
  unsigned short* wqbf = (unsigned short*)alloc((size_t)Cc*Cc*2);
  unsigned short* wkbf = (unsigned short*)alloc((size_t)Cc*Cc*2);
  unsigned short* wvbf = (unsigned short*)alloc((size_t)Cc*Cc*2);
  unsigned short* wgbf = (unsigned short*)alloc((size_t)Cc*Cc*2);
  unsigned short* wobf = (unsigned short*)alloc((size_t)Cc*Cc*2);
  float* pre   = (float*)alloc((size_t)BT*Cc*4);   // reused q/k/v pre-conv, then scan output
  float* qc    = (float*)alloc((size_t)BT*Cc*4);
  float* kc    = (float*)alloc((size_t)BT*Cc*4);
  float* vc    = (float*)alloc((size_t)BT*Cc*4);
  float* gbuf  = (float*)alloc((size_t)BT*Cc*4);
  float* alpha = (float*)alloc((size_t)BT*Hh*4);
  float* beta  = (float*)alloc((size_t)BT*Hh*4);
  float* obuf  = pre;                              // free after conv(v)
  unsigned short* obf = (unsigned short*)qc;       // free after scan

  const dim3 gemmGrid(Cc/64, BT/128);              // (16, 32)
  const int CAST_BLK = 256;
  const int castGridX  = (BT*Cc + CAST_BLK - 1) / CAST_BLK;
  const int castGridW  = (Cc*Cc + CAST_BLK - 1) / CAST_BLK;
  const int convGrid   = (BT*Cc + CAST_BLK - 1) / CAST_BLK;
  const int groupGrid  = (BT*Hh) / 8;              // 8 waves/block

  // casts
  cast_bf16_kernel<<<castGridX, CAST_BLK, 0, stream>>>(x,  xbf,  BT*Cc);
  cast_bf16_kernel<<<castGridW, CAST_BLK, 0, stream>>>(Wq, wqbf, Cc*Cc);
  cast_bf16_kernel<<<castGridW, CAST_BLK, 0, stream>>>(Wk, wkbf, Cc*Cc);
  cast_bf16_kernel<<<castGridW, CAST_BLK, 0, stream>>>(Wv, wvbf, Cc*Cc);
  cast_bf16_kernel<<<castGridW, CAST_BLK, 0, stream>>>(Wg, wgbf, Cc*Cc);
  cast_bf16_kernel<<<castGridW, CAST_BLK, 0, stream>>>(Wo, wobf, Cc*Cc);

  // q path
  gemm_bf16_kernel<<<gemmGrid, 256, 0, stream>>>(xbf, wqbf, pre, BT, Cc, Cc);
  conv_silu_kernel<<<convGrid, CAST_BLK, 0, stream>>>(pre, cqw, cqb, qc);
  // k path
  gemm_bf16_kernel<<<gemmGrid, 256, 0, stream>>>(xbf, wkbf, pre, BT, Cc, Cc);
  conv_silu_kernel<<<convGrid, CAST_BLK, 0, stream>>>(pre, ckw, ckb, kc);
  // v path
  gemm_bf16_kernel<<<gemmGrid, 256, 0, stream>>>(xbf, wvbf, pre, BT, Cc, Cc);
  conv_silu_kernel<<<convGrid, CAST_BLK, 0, stream>>>(pre, cvw, cvb, vc);
  // g path (no conv/activation)
  gemm_bf16_kernel<<<gemmGrid, 256, 0, stream>>>(xbf, wgbf, gbuf, BT, Cc, Cc);

  // rope + l2norm on q and k (in place)
  rope_l2norm_kernel<<<groupGrid, 256, 0, stream>>>(qc);
  rope_l2norm_kernel<<<groupGrid, 256, 0, stream>>>(kc);

  // alpha/beta gates from full-precision x
  proj_ab_kernel<<<BT, 64, 0, stream>>>(x, Wb, bb, Wgk, bgk, A_log, dt_bias, alpha, beta);

  // recurrent scan: 32 blocks (B*H), 64 threads each
  scan_kernel<<<Bb*Hh, 64, 0, stream>>>(qc, kc, vc, alpha, beta, Dp, obuf);

  // rmsnorm + silu + gate -> bf16
  post_kernel<<<groupGrid, 256, 0, stream>>>(obuf, gbuf, onw, obf);

  // final projection -> f32 output
  gemm_bf16_kernel<<<gemmGrid, 256, 0, stream>>>(obf, wobf, out, BT, Cc, Cc);
}
